// MaskedSelfAttention_3839700762720
// MI455X (gfx1250) — compile-verified, matrix operand order checked by reading, not run on hardware
//
#include <hip/hip_runtime.h>

// ---------------------------------------------------------------------------
// MI455X (gfx1250) masked self-attention, bf16 WMMA (v_wmma_f32_16x16x32_bf16)
//   B=8, S=2048, E=1024, fp32 in/out, bf16 tensor-core path, fp32 accumulate.
// Phase 0: fp32 -> bf16 pre-conversion of x and Wq/Wk/Wv (once, out of loop)
// Phase 1: QKV projection  (x @ W^T + b) -> Q,K bf16 [B*S,E], V^T bf16 [B,E,S]
// Phase 2: flash attention (online softmax) -> out fp32 [B,S,E]
// ---------------------------------------------------------------------------

#define BATCH   8
#define SEQ     2048
#define EMB     1024
#define NTOK    (BATCH * SEQ)          // 16384 tokens
#define SCALE   0.03125f               // 1/sqrt(1024)

typedef __bf16 bf16;
typedef __attribute__((ext_vector_type(16))) __bf16 v16bf;
typedef __attribute__((ext_vector_type(8)))  __bf16 v8bf;
typedef __attribute__((ext_vector_type(4)))  __bf16 v4bf;
typedef __attribute__((ext_vector_type(8)))  float  v8f;

// D = A(16x32 bf16) * B(32x16 bf16) + C(16x16 f32)
__device__ __forceinline__ v8f wmma_bf16(v16bf a, v16bf b, v8f c) {
  return __builtin_amdgcn_wmma_f32_16x16x32_bf16(
      /*neg_a=*/false, a, /*neg_b=*/false, b,
      /*c_mod=*/(short)0, c, /*reuse_a=*/false, /*reuse_b=*/false);
}

// A-tile loader (also B-tile: column n of B == row n of a row-major source).
// ISA 16-bit A layout: lane L holds source row (L&15); K values
//   lanes 0-15 : K = {0..7, 16..23},  lanes 16-31 : K = {8..15, 24..31}
__device__ __forceinline__ v16bf load_tile_bf16(const bf16* __restrict__ p,
                                                int ld, int lane) {
  const int r  = lane & 15;
  const int kb = (lane >> 4) << 3;
  const bf16* q = p + r * ld + kb;
  v8bf lo = *(const v8bf*)q;                 // 16B global_load_b128
  v8bf hi = *(const v8bf*)(q + 16);
  v16bf o;
#pragma unroll
  for (int i = 0; i < 8; ++i) { o[i] = lo[i]; o[8 + i] = hi[i]; }
  return o;
}

// ---------------------------------------------------------------------------
// Phase 0: bulk fp32 -> bf16 conversion (4 elements / thread)
// ---------------------------------------------------------------------------
__global__ __launch_bounds__(256) void cvt_bf16_kernel(
    const float* __restrict__ in, bf16* __restrict__ out, int n4) {
  int i = blockIdx.x * 256 + threadIdx.x;
  if (i < n4) {
    float4 v = ((const float4*)in)[i];
    v4bf o;
    o[0] = (__bf16)v.x; o[1] = (__bf16)v.y;
    o[2] = (__bf16)v.z; o[3] = (__bf16)v.w;
    ((v4bf*)out)[i] = o;
  }
}

// ---------------------------------------------------------------------------
// Phase 1: QKV projection from pre-converted bf16 operands.
//   One wave -> one (matrix, 16-row m-tile, 128-col n-slab) tile.
//   waves = 3 * (16384/16) * (1024/128) = 24576 ; 8 waves/block -> 3072 blocks
//   Inner loop: 18 x global_load_b128 + 8 x v_wmma, zero VALU.
// ---------------------------------------------------------------------------
__global__ __launch_bounds__(256) void qkv_proj_kernel(
    const bf16* __restrict__ xb,
    const bf16* __restrict__ Wqb, const float* __restrict__ bq,
    const bf16* __restrict__ Wkb, const float* __restrict__ bk,
    const bf16* __restrict__ Wvb, const float* __restrict__ bv,
    bf16* __restrict__ Qo, bf16* __restrict__ Ko, bf16* __restrict__ Vt) {
  const int lane = threadIdx.x & 31;
  const int wid  = blockIdx.x * 8 + (threadIdx.x >> 5);

  const int mat = wid >> 13;                 // 8192 tiles per matrix
  const int rem = wid & 8191;
  const int mt  = rem >> 3;                  // 1024 m-tiles of 16 tokens
  const int n8  = rem & 7;                   // 8 n-slabs of 128 cols

  const bf16*  W    = (mat == 0) ? Wqb : (mat == 1) ? Wkb : Wvb;
  const float* bias = (mat == 0) ? bq  : (mat == 1) ? bk  : bv;

  const int mrow  = mt * 16;
  const int nbase = n8 * 128;

  v8f acc[8] = {};                           // 16x128 fp32 accumulators
  for (int kk = 0; kk < 32; ++kk) {          // K = 1024 in steps of 32
    v16bf a = load_tile_bf16(xb + mrow * EMB + kk * 32, EMB, lane);
#pragma unroll
    for (int j = 0; j < 8; ++j) {
      // B column n = W row n (computing x @ W^T)
      v16bf bt = load_tile_bf16(W + (nbase + j * 16) * EMB + kk * 32, EMB, lane);
      acc[j] = wmma_bf16(a, bt, acc[j]);
    }
  }

  const int half8 = (lane >> 4) << 3;        // C/D layout: rows g / g+8
#pragma unroll
  for (int j = 0; j < 8; ++j) {
    const int col = nbase + j * 16 + (lane & 15);
    const float bcol = bias[col];
    if (mat < 2) {
      bf16* out = (mat == 0) ? Qo : Ko;      // row-major [NTOK, EMB]
#pragma unroll
      for (int g = 0; g < 8; ++g) {
        int m = mrow + g + half8;
        out[m * EMB + col] = (__bf16)(acc[j][g] + bcol);
      }
    } else {                                 // V stored transposed [B, E, S]
#pragma unroll
      for (int g = 0; g < 8; ++g) {
        int m  = mrow + g + half8;
        int bb = m >> 11, ss = m & (SEQ - 1);
        Vt[(size_t)bb * EMB * SEQ + (size_t)col * SEQ + ss] =
            (__bf16)(acc[j][g] + bcol);
      }
    }
  }
}

// ---------------------------------------------------------------------------
// Phase 2: flash attention. Block = (batch b, 16-row q-tile).
//   8 waves x 128 embedding cols each. 32-key chunks, causal-truncated.
// ---------------------------------------------------------------------------
__global__ __launch_bounds__(256) void attn_kernel(
    const bf16* __restrict__ Qo, const bf16* __restrict__ Ko,
    const bf16* __restrict__ Vt, float* __restrict__ out) {
  __shared__ float s_scores[16 * 32];        // 16 q-rows x 32 keys
  __shared__ float s_alpha[16];
  __shared__ float s_l[16];

  const int tid   = threadIdx.x;
  const int lane  = tid & 31;
  const int wave  = tid >> 5;
  const int qt    = blockIdx.x & 127;        // 128 q-tiles per sequence
  const int b     = blockIdx.x >> 7;
  const int qtok0 = b * SEQ + qt * 16;       // global token base
  const int ecol0 = wave * 128;              // this wave's embedding slice
  const int r     = lane & 15;
  const int kb    = (lane >> 4) << 3;        // 0 / 8 : half-wave offset
  const int qrow  = qt * 16 + r;             // this lane's softmax row

  // Cache this wave's Q slice (16 x 128) in registers as 4 A-tiles.
  v16bf qreg[4];
#pragma unroll
  for (int e4 = 0; e4 < 4; ++e4)
    qreg[e4] = load_tile_bf16(Qo + qtok0 * EMB + ecol0 + e4 * 32, EMB, lane);

  v8f acc[8] = {};                           // 16 x 128 fp32 output accum
  float mcur = -1e30f, lcur = 0.0f;

  const int nchunks = (qt * 16 + 15) / 32 + 1;
  for (int cb = 0; cb < nchunks; ++cb) {
    const int k0 = cb * 32;
    __syncthreads();                         // protect LDS reuse
    s_scores[tid] = 0.0f; s_scores[tid + 256] = 0.0f;
    __syncthreads();

    // --- partial Q*K^T over this wave's e-slice, reduce via LDS atomics ---
#pragma unroll
    for (int t = 0; t < 2; ++t) {            // two 16-key sub-tiles
      v8f sa = {};
      const bf16* kbase = Ko + (b * SEQ + k0 + t * 16) * EMB + ecol0;
#pragma unroll
      for (int e4 = 0; e4 < 4; ++e4)
        sa = wmma_bf16(qreg[e4], load_tile_bf16(kbase + e4 * 32, EMB, lane), sa);
#pragma unroll
      for (int g = 0; g < 8; ++g)            // D row = g + kb, col = lane&15
        atomicAdd(&s_scores[(g + kb) * 32 + t * 16 + (lane & 15)], sa[g]);
    }
    __syncthreads();

    // --- online softmax over the 32 masked+scaled scores of row r ---
    float cmax = -1e30f;
#pragma unroll
    for (int c = 0; c < 32; ++c) {
      float s = s_scores[r * 32 + c] * SCALE;
      s = (k0 + c > qrow) ? -1e9f : s;       // exact reference masking
      cmax = fmaxf(cmax, s);
    }
    const float mnew  = fmaxf(mcur, cmax);
    const float alpha = __expf(mcur - mnew);
    float psum = 0.0f;
#pragma unroll
    for (int c = 0; c < 32; ++c) {
      float s = s_scores[r * 32 + c] * SCALE;
      s = (k0 + c > qrow) ? -1e9f : s;
      psum += __expf(s - mnew);
    }
    lcur = lcur * alpha + psum;
    mcur = mnew;
    if (wave == 0 && lane < 16) s_alpha[r] = alpha;  // broadcast per-row alpha
    __syncthreads();

    // --- rescale accumulators (C layout rows g+kb) ---
    float a8[8];
#pragma unroll
    for (int g = 0; g < 8; ++g) a8[g] = s_alpha[g + kb];
#pragma unroll
    for (int j = 0; j < 8; ++j)
#pragma unroll
      for (int g = 0; g < 8; ++g) acc[j][g] *= a8[g];

    // --- build P (16x32 bf16) in A-tile layout: K cols {kb..kb+7, kb+16..} ---
    v16bf P;
#pragma unroll
    for (int i = 0; i < 8; ++i) {
      int c = kb + i;
      float s = s_scores[r * 32 + c] * SCALE;
      s = (k0 + c > qrow) ? -1e9f : s;
      P[i] = (__bf16)__expf(s - mnew);
    }
#pragma unroll
    for (int i = 0; i < 8; ++i) {
      int c = kb + 16 + i;
      float s = s_scores[r * 32 + c] * SCALE;
      s = (k0 + c > qrow) ? -1e9f : s;
      P[8 + i] = (__bf16)__expf(s - mnew);
    }

    // --- acc += P * V  (B column e = row e of V^T, contiguous) ---
    const bf16* vbase = Vt + (size_t)b * EMB * SEQ + k0;
#pragma unroll
    for (int j = 0; j < 8; ++j) {
      v16bf vt = load_tile_bf16(vbase + (size_t)(ecol0 + j * 16) * SEQ, SEQ, lane);
      acc[j] = wmma_bf16(P, vt, acc[j]);
    }
  }

  // --- final normalization and fp32 store ---
  __syncthreads();
  if (wave == 0 && lane < 16) s_l[r] = lcur;
  __syncthreads();
#pragma unroll
  for (int g = 0; g < 8; ++g) {
    const float inv = 1.0f / s_l[g + kb];
    const int m = qtok0 + g + kb;
#pragma unroll
    for (int j = 0; j < 8; ++j) {
      int col = ecol0 + j * 16 + (lane & 15);
      out[(size_t)m * EMB + col] = acc[j][g] * inv;
    }
  }
}

// ---------------------------------------------------------------------------
extern "C" void kernel_launch(void* const* d_in, const int* in_sizes, int n_in,
                              void* d_out, int out_size, void* d_ws,
                              size_t ws_size, hipStream_t stream) {
  const float* x  = (const float*)d_in[0];
  const float* Wq = (const float*)d_in[1];
  const float* bq = (const float*)d_in[2];
  const float* Wk = (const float*)d_in[3];
  const float* bk = (const float*)d_in[4];
  const float* Wv = (const float*)d_in[5];
  const float* bv = (const float*)d_in[6];
  float* out = (float*)d_out;

  // Workspace layout (bf16): Q | K | V^T | x_bf16 | Wq_bf16 | Wk_bf16 | Wv_bf16
  const size_t NE = (size_t)NTOK * EMB;      // 16,777,216 elements
  const size_t WE = (size_t)EMB * EMB;       //  1,048,576 elements
  bf16* Qo  = (bf16*)d_ws;
  bf16* Ko  = Qo  + NE;
  bf16* Vt  = Ko  + NE;
  bf16* xb  = Vt  + NE;
  bf16* Wqb = xb  + NE;
  bf16* Wkb = Wqb + WE;
  bf16* Wvb = Wkb + WE;

  // Phase 0: one-time fp32 -> bf16 conversion (removes all cvt from GEMM loop)
  cvt_bf16_kernel<<<(int)(NE / 4 / 256), 256, 0, stream>>>(x,  xb,  (int)(NE / 4));
  cvt_bf16_kernel<<<(int)(WE / 4 / 256), 256, 0, stream>>>(Wq, Wqb, (int)(WE / 4));
  cvt_bf16_kernel<<<(int)(WE / 4 / 256), 256, 0, stream>>>(Wk, Wkb, (int)(WE / 4));
  cvt_bf16_kernel<<<(int)(WE / 4 / 256), 256, 0, stream>>>(Wv, Wvb, (int)(WE / 4));

  // Phase 1: 3 mats * 1024 m-tiles * 8 n-slabs = 24576 waves / 8 per block
  qkv_proj_kernel<<<3072, 256, 0, stream>>>(xb, Wqb, bq, Wkb, bk, Wvb, bv,
                                            Qo, Ko, Vt);
  // Phase 2: one block per (batch, 16-row q-tile)
  attn_kernel<<<BATCH * (SEQ / 16), 256, 0, stream>>>(Qo, Ko, Vt, out);
}
